// NMNISTNet_82738249990687
// MI455X (gfx1250) — matrix operand order repaired and checked by reading, not compiled, not run on hardware
//
#include <hip/hip_runtime.h>

typedef __attribute__((ext_vector_type(16))) __bf16 v16bf;
typedef __attribute__((ext_vector_type(8)))  __bf16 v8bf;
typedef __attribute__((ext_vector_type(8)))  float  v8f;

#define THETA 10.0f

// -----------------------------------------------------------------------------
// Batched GEMM via CDNA5 WMMA (bf16 inputs, f32 accumulate), double-buffered.
//   Out[b] (MxN, f32, row-major stride N) = W (MxK, f32) * In[b] (KxN, f32)
// Block: 128 threads (4 wave32). Block tile 64(M) x 32(N), K stepped by 32.
// Each wave: one 16-row A fragment, two N-subtiles -> 2 WMMAs per K-step.
// Pipeline: global loads for chunk k+1 issue before the WMMAs of chunk k;
// LDS is double-buffered so only ONE barrier per K-step is needed.
// Requires K % 4 == 0 and N % 2 == 0 (true here: K in {2312,500}, N = 350).
// -----------------------------------------------------------------------------
__global__ __launch_bounds__(128)
void gemm_wmma_bf16(const float* __restrict__ W, const float* __restrict__ In,
                    float* __restrict__ Out,
                    int M, int K, int N,
                    long long inBatchStride, long long outBatchStride)
{
    __shared__ __attribute__((aligned(16))) __bf16 As[2][64 * 32];  // [m][k]
    __shared__ __attribute__((aligned(16))) __bf16 Bs[2][32 * 32];  // [n][k]

    const int tid  = threadIdx.x;
    const int lane = tid & 31;
    const int wave = tid >> 5;

    const int n0 = blockIdx.x * 32;
    const int m0 = blockIdx.y * 64;

    const float* __restrict__ inB  = In  + (long long)blockIdx.z * inBatchStride;
    float*       __restrict__ outB = Out + (long long)blockIdx.z * outBatchStride;

    // per-thread in-flight staging registers (distinct regs -> loads batch up)
    float4 aV[4]; float2 bV[4]; bool aOk[4], bOk[4];

    // issue all 8 global loads for chunk kc (addresses clamped; no branches)
    auto stage_load = [&](int kc) {
        #pragma unroll
        for (int i = 0; i < 4; ++i) {                      // A: 64x32 as 512 float4s
            const int f4 = tid + i * 128;
            const int r  = f4 >> 3, c4 = (f4 & 7) << 2;
            const int gm = m0 + r, gk = kc + c4;
            aOk[i] = (gm < M) && (gk < K);                 // K%4==0 => quad valid
            const long long off = aOk[i] ? ((long long)gm * K + gk) : 0;
            aV[i] = *(const float4*)(W + off);
        }
        #pragma unroll
        for (int i = 0; i < 4; ++i) {                      // B: 32x32 as 512 float2s
            const int f2 = tid + i * 128;
            const int r  = f2 >> 4, c2 = (f2 & 15) << 1;
            const int gk = kc + r, gn = n0 + c2;
            bOk[i] = (gk < K) && (gn < N);                 // N even => pair valid
            const long long off = bOk[i] ? ((long long)gk * N + gn) : 0;
            bV[i] = *(const float2*)(inB + off);
        }
    };
    // convert f32 -> bf16 and commit staged registers into LDS buffer `buf`
    auto stage_store = [&](int buf) {
        #pragma unroll
        for (int i = 0; i < 4; ++i) {
            const int f4 = tid + i * 128;
            const int r  = f4 >> 3, c4 = (f4 & 7) << 2;
            float4 v = aV[i];
            v.x = aOk[i] ? v.x : 0.0f;  v.y = aOk[i] ? v.y : 0.0f;
            v.z = aOk[i] ? v.z : 0.0f;  v.w = aOk[i] ? v.w : 0.0f;
            __bf16* dst = &As[buf][r * 32 + c4];           // 8B-aligned
            dst[0] = (__bf16)v.x; dst[1] = (__bf16)v.y;
            dst[2] = (__bf16)v.z; dst[3] = (__bf16)v.w;
        }
        #pragma unroll
        for (int i = 0; i < 4; ++i) {
            const int f2 = tid + i * 128;
            const int r  = f2 >> 4, c2 = (f2 & 15) << 1;
            float2 v = bV[i];
            v.x = bOk[i] ? v.x : 0.0f;  v.y = bOk[i] ? v.y : 0.0f;
            Bs[buf][(c2    ) * 32 + r] = (__bf16)v.x;      // transposed [n][k]
            Bs[buf][(c2 + 1) * 32 + r] = (__bf16)v.y;
        }
    };

    v8f acc0 = {}, acc1 = {};

    // ---- pipeline prologue: chunk 0 into LDS[0] ----
    stage_load(0);
    stage_store(0);
    __syncthreads();

    // fragment coordinates (CDNA5 bf16 WMMA register layout)
    const int row   = (wave << 4) + (lane & 15);
    const int klo   = (lane < 16) ? 0 : 8;
    const int col   = lane & 15;
    const int kbase = (lane < 16) ? 0 : 8;

    int buf = 0;
    for (int kc = 0; kc < K; kc += 32) {
        const bool more = (kc + 32) < K;
        if (more) stage_load(kc + 32);                     // overlap with math below

        const v8bf alo = *(const v8bf*)&As[buf][row * 32 + klo];
        const v8bf ahi = *(const v8bf*)&As[buf][row * 32 + klo + 16];
        v16bf af;
        #pragma unroll
        for (int i = 0; i < 8; ++i) { af[i] = alo[i]; af[i + 8] = ahi[i]; }

        {
            const v8bf blo = *(const v8bf*)&Bs[buf][col * 32 + kbase];
            const v8bf bhi = *(const v8bf*)&Bs[buf][col * 32 + kbase + 16];
            v16bf bfr;
            #pragma unroll
            for (int i = 0; i < 8; ++i) { bfr[i] = blo[i]; bfr[i + 8] = bhi[i]; }
            acc0 = __builtin_amdgcn_wmma_f32_16x16x32_bf16(false, af, false, bfr,
                                                           (short)0, acc0, false, false);
        }
        {
            const v8bf blo = *(const v8bf*)&Bs[buf][(16 + col) * 32 + kbase];
            const v8bf bhi = *(const v8bf*)&Bs[buf][(16 + col) * 32 + kbase + 16];
            v16bf bfr;
            #pragma unroll
            for (int i = 0; i < 8; ++i) { bfr[i] = blo[i]; bfr[i + 8] = bhi[i]; }
            acc1 = __builtin_amdgcn_wmma_f32_16x16x32_bf16(false, af, false, bfr,
                                                           (short)0, acc1, false, false);
        }

        if (more) stage_store(buf ^ 1);                    // commit next chunk
        __syncthreads();                                   // single barrier per step
        buf ^= 1;
    }

    // D layout: VGPR r -> M = r + (lane<16 ? 0 : 8), N = lane&15
    const int mbase = m0 + (wave << 4) + ((lane < 16) ? 0 : 8);
    #pragma unroll
    for (int sub = 0; sub < 2; ++sub) {
        const int gcol = n0 + sub * 16 + col;
        if (gcol < N) {
            const v8f& a = sub ? acc1 : acc0;
            #pragma unroll
            for (int r = 0; r < 8; ++r) {
                const int gm = mbase + r;
                if (gm < M) outB[(long long)gm * N + gcol] = a[r];
            }
        }
    }
}

// -----------------------------------------------------------------------------
// Causal FIR along time:  out[row,t] = sum_{j=0..klen-1, j<=t} kern[j]*in[row,t-j]
// One block per (batch,feature) row; row + kernel staged in LDS. (t_s == 1)
// -----------------------------------------------------------------------------
__global__ __launch_bounds__(256)
void causal_conv(const float* __restrict__ in, const float* __restrict__ kern,
                 int klen, float* __restrict__ out, int T)
{
    __shared__ float sk[160];
    __shared__ float srow[352];          // T = 350 <= 352

    const long long row = blockIdx.x;
    const float* __restrict__ ir   = in  + row * T;
    float*       __restrict__ orow = out + row * T;

    const int kc = (klen < 160) ? klen : 160;
    for (int i = threadIdx.x; i < kc; i += blockDim.x) sk[i] = kern[i];
    for (int i = threadIdx.x; i < T;  i += blockDim.x) srow[i] = ir[i];
    __syncthreads();

    for (int t = threadIdx.x; t < T; t += blockDim.x) {
        float a = 0.0f;
        const int jm = (kc - 1 < t) ? (kc - 1) : t;
        for (int j = 0; j <= jm; ++j) a += sk[j] * srow[t - j];
        orow[t] = a;
    }
}

// -----------------------------------------------------------------------------
// Sequential refractory spike scan. One thread per (batch,feature) row.
// Ring buffer of pending refractory contributions in LDS (64 slots >= reflen).
//   u[t] = a[t] + ring[head]; spike -> add ref[1..L-1] to future slots.
// -----------------------------------------------------------------------------
__global__ __launch_bounds__(128)
void spike_scan(const float* __restrict__ a, const float* __restrict__ ref,
                int reflen, float* __restrict__ s, int T, long long rows)
{
    __shared__ float ring[128 * 64];
    __shared__ float sref[64];

    const int tid = threadIdx.x;
    const int L   = (reflen < 64) ? reflen : 64;
    for (int i = tid; i < L; i += blockDim.x) sref[i] = ref[i];
    float* rb = &ring[tid * 64];
    #pragma unroll 4
    for (int i = 0; i < 64; ++i) rb[i] = 0.0f;
    __syncthreads();

    const long long row = (long long)blockIdx.x * blockDim.x + tid;
    if (row >= rows) return;

    const float* __restrict__ ar = a + row * T;
    float*       __restrict__ sr = s + row * T;

    int head = 0;
    for (int t = 0; t < T; ++t) {
        const float u  = ar[t] + rb[head];
        const float sp = (u >= THETA) ? 1.0f : 0.0f;   // / t_s, t_s == 1
        sr[t] = sp;
        if (sp != 0.0f) {                               // ref[0] == 0, skip j=0
            for (int j = 1; j < L; ++j) rb[(head + j) & 63] += sref[j];
        }
        rb[head] = 0.0f;                                // consumed slot recycles
        head = (head + 1) & 63;
    }
}

// -----------------------------------------------------------------------------
extern "C" void kernel_launch(void* const* d_in, const int* in_sizes, int n_in,
                              void* d_out, int out_size, void* d_ws, size_t ws_size,
                              hipStream_t stream) {
    (void)n_in; (void)out_size; (void)ws_size;
    const float* x   = (const float*)d_in[0];  // (B, C*X*Y, T) = (32, 2312, 350)
    const float* W1  = (const float*)d_in[1];  // (500, 2312)
    const float* W2  = (const float*)d_in[2];  // (500, 500)
    const float* W3  = (const float*)d_in[3];  // (10, 500)
    const float* srm = (const float*)d_in[4];
    const float* ref = (const float*)d_in[5];
    const int Ksrm = in_sizes[4];
    const int Kref = in_sizes[5];

    const int B = 32, NIN = 2 * 34 * 34, NH = 500, NO = 10, T = 350;
    float* out = (float*)d_out;                // (32, 10, 350)

    const size_t per = (size_t)B * NH * T;     // 5.6M f32 per buffer
    float* buf0 = (float*)d_ws;                // gemm output
    float* buf1 = buf0 + per;                  // conv output
    float* buf2 = buf1 + per;                  // spike output (next gemm input)

    const dim3 gblk(128), cblk(256), sblk(128);

    // ---- layer 1: h = W1 @ x ; a = srm * h ; s1 = spike(a)   (conv<->matmul commuted)
    {
        dim3 g((T + 31) / 32, (NH + 63) / 64, B);
        gemm_wmma_bf16<<<g, gblk, 0, stream>>>(W1, x, buf0, NH, NIN, T,
                                               (long long)NIN * T, (long long)NH * T);
        causal_conv<<<dim3(B * NH), cblk, 0, stream>>>(buf0, srm, Ksrm, buf1, T);
        long long rows = (long long)B * NH;
        spike_scan<<<dim3((unsigned)((rows + 127) / 128)), sblk, 0, stream>>>(
            buf1, ref, Kref, buf2, T, rows);
    }
    // ---- layer 2
    {
        dim3 g((T + 31) / 32, (NH + 63) / 64, B);
        gemm_wmma_bf16<<<g, gblk, 0, stream>>>(W2, buf2, buf0, NH, NH, T,
                                               (long long)NH * T, (long long)NH * T);
        causal_conv<<<dim3(B * NH), cblk, 0, stream>>>(buf0, srm, Ksrm, buf1, T);
        long long rows = (long long)B * NH;
        spike_scan<<<dim3((unsigned)((rows + 127) / 128)), sblk, 0, stream>>>(
            buf1, ref, Kref, buf2, T, rows);
    }
    // ---- layer 3 (writes spikes straight to d_out)
    {
        dim3 g((T + 31) / 32, 1, B);
        gemm_wmma_bf16<<<g, gblk, 0, stream>>>(W3, buf2, buf0, NO, NH, T,
                                               (long long)NH * T, (long long)NO * T);
        causal_conv<<<dim3(B * NO), cblk, 0, stream>>>(buf0, srm, Ksrm, buf1, T);
        long long rows = (long long)B * NO;
        spike_scan<<<dim3((unsigned)((rows + 127) / 128)), sblk, 0, stream>>>(
            buf1, ref, Kref, out, T, rows);
    }
}
